// LinearAttention_32220844655288
// MI455X (gfx1250) — matrix operand order repaired and checked by reading, not compiled, not run on hardware
//
#include <hip/hip_runtime.h>

#define B_  4
#define L_  4096
#define D_  1024
#define H_  16
#define HD_ 64
#define BH_ (B_*H_)

typedef _Float16 half_t;
typedef __attribute__((ext_vector_type(16))) _Float16 v16h;
typedef __attribute__((ext_vector_type(8)))  float    v8f;

union HU { int4 q[2]; v16h v; };

static __device__ inline v8f wmma_f16(v16h a, v16h b, v8f c) {
    // v_wmma_f32_16x16x32_f16: (neg_a, A, neg_b, B, c_mod, C, reuse_a, reuse_b)
    return __builtin_amdgcn_wmma_f32_16x16x32_f16(false, a, false, b, (short)0, c,
                                                  false, false);
}

// ---------------------------------------------------------------------------
// Stage 1: fused QKV projection + feature map.
//   Y = X @ W^T + b ; q,k: elu(y)+1 = max(y,0)+exp(min(y,0)) ; k *= mask.
//   Output f16 [B,H,L,HD]. Block: 128 threads (4 waves), M=64, N=64, Kstep=32.
//   Double-buffered LDS W-tile + register prefetch of next tile.
// ---------------------------------------------------------------------------
__global__ __launch_bounds__(128)
void qkv_kernel(const float* __restrict__ X,
                const float* __restrict__ Wq, const float* __restrict__ bq,
                const float* __restrict__ Wk, const float* __restrict__ bk,
                const float* __restrict__ Wv, const float* __restrict__ bv,
                const float* __restrict__ mask,
                half_t* __restrict__ qf, half_t* __restrict__ kf,
                half_t* __restrict__ vf)
{
    const int mode = blockIdx.z;                       // 0=q 1=k 2=v
    const float* W    = (mode == 0) ? Wq : (mode == 1) ? Wk : Wv;
    const float* bias = (mode == 0) ? bq : (mode == 1) ? bk : bv;
    half_t*      out  = (mode == 0) ? qf : (mode == 1) ? kf : vf;

    __shared__ half_t Wt[2][64 * 32];                  // [n][k], k contiguous

    const int tid   = threadIdx.x;
    const int lane  = tid & 31;
    const int wave  = tid >> 5;
    const int col16 = lane & 15;
    const int hi    = (lane >> 4) & 1;
    const int o1    = hi ? 8 : 0;
    const int kof   = hi ? 16 : 0;

    const int mbase = blockIdx.x * 64;                 // over B*L
    const int nbase = blockIdx.y * 64;                 // over D
    const int arow  = mbase + wave * 16 + col16;       // A-fragment row

    // cooperative W-tile addressing: thread covers row wn, cols wk0..wk0+15
    const int e   = tid * 16;
    const int wn  = e >> 5, wk0 = e & 31;
    const float* wsrc = W + (size_t)(nbase + wn) * D_ + wk0;
    const float* xr   = X + (size_t)arow * D_;

    auto loadW = [&](int i, float* w) {
        const float* s = wsrc + i * 32;
        *(float4*)&w[0]  = *(const float4*)(s + 0);
        *(float4*)&w[4]  = *(const float4*)(s + 4);
        *(float4*)&w[8]  = *(const float4*)(s + 8);
        *(float4*)&w[12] = *(const float4*)(s + 12);
    };
    auto loadA = [&](int i, float* a) {
        const float* s = xr + i * 32 + o1;
        *(float4*)&a[0]  = *(const float4*)(s + 0);
        *(float4*)&a[4]  = *(const float4*)(s + 4);
        *(float4*)&a[8]  = *(const float4*)(s + 16);
        *(float4*)&a[12] = *(const float4*)(s + 20);
    };

    v8f acc[4];
#pragma unroll
    for (int t = 0; t < 4; t++)
#pragma unroll
        for (int r = 0; r < 8; r++) acc[t][r] = 0.f;

    float wreg[16], areg[16];
    loadW(0, wreg);
    loadA(0, areg);

    for (int i = 0; i < 32; i++) {                     // K loop, Kstep=32
        half_t* wt = &Wt[i & 1][0];
        // commit prefetched W tile to LDS (f32 -> f16)
#pragma unroll
        for (int j = 0; j < 16; j++) wt[wn * 32 + wk0 + j] = (half_t)wreg[j];
        // convert prefetched A regs to fragment
        v16h a;
#pragma unroll
        for (int j = 0; j < 16; j++) a[j] = (half_t)areg[j];
        // issue next tile's global loads (latency hidden behind WMMA below)
        if (i + 1 < 32) { loadW(i + 1, wreg); loadA(i + 1, areg); }
        __syncthreads();
#pragma unroll
        for (int t = 0; t < 4; t++) {                  // 4 N-tiles of 16
            HU ub;
            const int n = t * 16 + col16;
            ub.q[0] = *(const int4*)&wt[n * 32 + o1];
            ub.q[1] = *(const int4*)&wt[n * 32 + o1 + 16];
            acc[t] = wmma_f16(a, ub.v, acc[t]);
        }
    }

    // epilogue: bias + branchless feature map + scatter to [B,H,L,HD] f16
    float mk[8];
    if (mode == 1) {
#pragma unroll
        for (int r = 0; r < 8; r++) mk[r] = mask[mbase + wave * 16 + r + hi * 8];
    }
#pragma unroll
    for (int t = 0; t < 4; t++) {
        const int   nglob = nbase + t * 16 + col16;
        const float bb    = bias[nglob];
        const int   h     = nglob >> 6, nn = nglob & 63;
#pragma unroll
        for (int r = 0; r < 8; r++) {
            const int row = mbase + wave * 16 + r + hi * 8;
            float y = acc[t][r] + bb;
            if (mode != 2) y = fmaxf(y, 0.f) + __expf(fminf(y, 0.f));   // elu+1
            if (mode == 1) y *= mk[r];
            const int b = row >> 12, l = row & (L_ - 1);
            out[(((size_t)b * H_ + h) * L_ + l) * HD_ + nn] = (half_t)y;
        }
    }
}

// ---------------------------------------------------------------------------
// Stage 2: kv[d,e] = sum_s K[s,d] V[s,e], ksum[d] = sum_s K[s,d] per (b,h).
//   grid (4 L-chunks, 64 bh), 128 threads. Wave w owns e-tile w; ksum via an
//   extra WMMA against a ones-column B fragment. Double-buffered transposed
//   LDS tiles; f32 global atomic adds into [64][65] (kv | ksum), L2-resident.
// ---------------------------------------------------------------------------
__global__ __launch_bounds__(128)
void kv_kernel(const half_t* __restrict__ kf, const half_t* __restrict__ vf,
               float* __restrict__ kvf)               // [BH][64][65], zeroed
{
    __shared__ half_t KtT[2][64 * 32];                // transposed: [d][s]
    __shared__ half_t VtT[2][64 * 32];                // transposed: [e][s]

    const int chunk = blockIdx.x;                     // 0..3
    const int bh    = blockIdx.y;                     // 0..63
    const int tid = threadIdx.x, lane = tid & 31, wave = tid >> 5;
    const int col16 = lane & 15, hi = (lane >> 4) & 1;
    const int o1 = hi ? 8 : 0, kof = hi ? 16 : 0;

    const half_t* Kb = kf + (size_t)bh * L_ * HD_;
    const half_t* Vb = vf + (size_t)bh * L_ * HD_;

    const int s0 = chunk * (L_ / 4);
    const int sl = (tid * 16) >> 6;                   // this thread's s row (0..31)
    const int d0 = (tid * 16) & 63;                   // and d column group

    auto loadKV = [&](int i, HU& uk, HU& uv) {
        const half_t* kp = Kb + (size_t)(s0 + i * 32 + sl) * HD_ + d0;
        const half_t* vp = Vb + (size_t)(s0 + i * 32 + sl) * HD_ + d0;
        uk.q[0] = *(const int4*)kp;  uk.q[1] = *(const int4*)(kp + 8);
        uv.q[0] = *(const int4*)vp;  uv.q[1] = *(const int4*)(vp + 8);
    };

    v8f acc[4], accS;
#pragma unroll
    for (int t = 0; t < 4; t++)
#pragma unroll
        for (int r = 0; r < 8; r++) acc[t][r] = 0.f;
#pragma unroll
    for (int r = 0; r < 8; r++) accS[r] = 0.f;

    v16h bones;                                       // B with ones in col n==0
#pragma unroll
    for (int j = 0; j < 16; j++) bones[j] = (col16 == 0) ? (half_t)1.0f : (half_t)0.0f;

    HU uk, uv;
    loadKV(0, uk, uv);

    for (int i = 0; i < 32; i++) {                    // 32 s-steps of 32
        const int buf = i & 1;
        // commit prefetched rows, transposed, to LDS
#pragma unroll
        for (int j = 0; j < 16; j++) {
            KtT[buf][(d0 + j) * 32 + sl] = uk.v[j];
            VtT[buf][(d0 + j) * 32 + sl] = uv.v[j];
        }
        if (i + 1 < 32) loadKV(i + 1, uk, uv);        // prefetch next tile
        __syncthreads();

        HU ub;                                        // B frag: this wave's e-tile
        {
            const int n = wave * 16 + col16;
            ub.q[0] = *(const int4*)&VtT[buf][n * 32 + o1];
            ub.q[1] = *(const int4*)&VtT[buf][n * 32 + o1 + 16];
        }
#pragma unroll
        for (int mt = 0; mt < 4; mt++) {              // A frag: K^T (d rows)
            HU ua;
            const int m = mt * 16 + col16;
            ua.q[0] = *(const int4*)&KtT[buf][m * 32 + o1];
            ua.q[1] = *(const int4*)&KtT[buf][m * 32 + o1 + 16];
            acc[mt] = wmma_f16(ua.v, ub.v, acc[mt]);
            if (mt == wave) accS = wmma_f16(ua.v, bones, accS);  // ksum rows
        }
    }

#pragma unroll
    for (int mt = 0; mt < 4; mt++)
#pragma unroll
        for (int r = 0; r < 8; r++) {
            const int m = mt * 16 + r + hi * 8;
            const int n = wave * 16 + col16;
            unsafeAtomicAdd(&kvf[(size_t)bh * 4160 + m * 65 + n], acc[mt][r]);
        }
    if (col16 == 0) {
#pragma unroll
        for (int r = 0; r < 8; r++) {
            const int m = wave * 16 + r + hi * 8;
            unsafeAtomicAdd(&kvf[(size_t)bh * 4160 + m * 65 + 64], accS[r]);
        }
    }
}

// ---------------------------------------------------------------------------
// Stage 3: ctx[l,e] = (sum_d q[l,d] kv[d,e]) * z[l],  z[l] = q[l,:]·ksum.
//   grid (32 l-tiles, 64 bh), 256 threads (8 waves x 16 rows), N=64, K=64.
//   z computed by WMMA against a ksum-in-column-0 B fragment.
// ---------------------------------------------------------------------------
__global__ __launch_bounds__(256)
void ctx_kernel(const half_t* __restrict__ qf, const float* __restrict__ kvf,
                float* __restrict__ out)
{
    __shared__ half_t KVT[64 * 64];                   // transposed kv: [e][d]
    __shared__ float  ks_s[64];
    __shared__ float  z_s[8 * 16];

    const int ltile = blockIdx.x;                     // 0..31
    const int bh    = blockIdx.y;                     // 0..63
    const int tid = threadIdx.x, lane = tid & 31, wave = tid >> 5;
    const int col16 = lane & 15, hi = (lane >> 4) & 1;
    const int o1 = hi ? 8 : 0, kof = hi ? 16 : 0;

    const float* kvb = kvf + (size_t)bh * 4160;
#pragma unroll
    for (int it = 0; it < 16; it++) {                 // stage kv f32->f16, transposed
        const int i = tid + it * 256;                 // 0..4095
        const int d = i >> 6, ee = i & 63;
        KVT[ee * 64 + d] = (half_t)kvb[d * 65 + ee];
    }
    if (tid < 64) ks_s[tid] = kvb[tid * 65 + 64];
    __syncthreads();

    const int lbase = ltile * 128 + wave * 16;
    const half_t* Qb = qf + (size_t)bh * L_ * HD_;

    HU ua0, ua1;                                      // Q fragments, 2 K-steps
    {
        const half_t* qr = Qb + (size_t)(lbase + col16) * HD_;
        ua0.q[0] = *(const int4*)(qr + kof);
        ua0.q[1] = *(const int4*)(qr + kof + 8);
        ua1.q[0] = *(const int4*)(qr + 32 + kof);
        ua1.q[1] = *(const int4*)(qr + 32 + kof + 8);
    }

    v8f acc[4];
#pragma unroll
    for (int t = 0; t < 4; t++)
#pragma unroll
        for (int r = 0; r < 8; r++) acc[t][r] = 0.f;

#pragma unroll
    for (int t = 0; t < 4; t++) {
        HU ub0, ub1;
        const int n = t * 16 + col16;
        ub0.q[0] = *(const int4*)&KVT[n * 64 + kof];
        ub0.q[1] = *(const int4*)&KVT[n * 64 + kof + 8];
        ub1.q[0] = *(const int4*)&KVT[n * 64 + 32 + kof];
        ub1.q[1] = *(const int4*)&KVT[n * 64 + 32 + kof + 8];
        acc[t] = wmma_f16(ua0.v, ub0.v, acc[t]);
        acc[t] = wmma_f16(ua1.v, ub1.v, acc[t]);
    }

    // z via WMMA: B fragment carrying ksum in column n==0 only
    {
        v16h bz0, bz1;
#pragma unroll
        for (int j = 0; j < 16; j++) {
            bz0[j] = (col16 == 0) ? (half_t)ks_s[kof + j]      : (half_t)0.0f;
            bz1[j] = (col16 == 0) ? (half_t)ks_s[32 + kof + j] : (half_t)0.0f;
        }
        v8f accZ;
#pragma unroll
        for (int r = 0; r < 8; r++) accZ[r] = 0.f;
        accZ = wmma_f16(ua0.v, bz0, accZ);
        accZ = wmma_f16(ua1.v, bz1, accZ);
        if (col16 == 0) {                             // lanes holding column 0
#pragma unroll
            for (int r = 0; r < 8; r++) z_s[wave * 16 + r + hi * 8] = accZ[r];
        }
    }
    __syncthreads();

    const int b = bh >> 4, h = bh & 15;
#pragma unroll
    for (int t = 0; t < 4; t++)
#pragma unroll
        for (int r = 0; r < 8; r++) {
            const int m = r + hi * 8;
            const int l = lbase + m;
            const int n = t * 16 + col16;
            out[((size_t)b * L_ + l) * D_ + h * HD_ + n] = acc[t][r] * z_s[wave * 16 + m];
        }
}

// ---------------------------------------------------------------------------
extern "C" void kernel_launch(void* const* d_in, const int* in_sizes, int n_in,
                              void* d_out, int out_size, void* d_ws, size_t ws_size,
                              hipStream_t stream)
{
    const float* X    = (const float*)d_in[0];
    const float* mask = (const float*)d_in[1];
    const float* Wq   = (const float*)d_in[2];
    const float* bq   = (const float*)d_in[3];
    const float* Wk   = (const float*)d_in[4];
    const float* bk   = (const float*)d_in[5];
    const float* Wv   = (const float*)d_in[6];
    const float* bv   = (const float*)d_in[7];
    float* out = (float*)d_out;

    const size_t nqkv = (size_t)BH_ * L_ * HD_;       // 16.7M f16 each
    half_t* qf  = (half_t*)d_ws;
    half_t* kf  = qf + nqkv;
    half_t* vf  = kf + nqkv;
    float*  kvf = (float*)(vf + nqkv);                // [BH][64][65]

    hipMemsetAsync(kvf, 0, (size_t)BH_ * 4160 * sizeof(float), stream);

    qkv_kernel<<<dim3(256, 16, 3), 128, 0, stream>>>(X, Wq, bq, Wk, bk, Wv, bv,
                                                     mask, qf, kf, vf);
    kv_kernel<<<dim3(4, 64), 128, 0, stream>>>(kf, vf, kvf);
    ctx_kernel<<<dim3(32, 64), 256, 0, stream>>>(qf, kvf, out);
}
